// DFlashAttentionV7_72292889526416
// MI455X (gfx1250) — compile-verified, matrix operand order checked by reading, not run on hardware
//
#include <hip/hip_runtime.h>

#define H_   4096
#define NH_  32
#define NKV_ 8
#define HD_  128
#define G_   4
#define B_   4
#define KQ_  64
#define CTX_ 4096
#define L_   4160              // CTX + KQ
#define DH_  (NH_ * HD_)       // 4096
#define DKV_ (NKV_ * HD_)      // 1024

typedef __attribute__((ext_vector_type(16))) __bf16 v16bf;
typedef __attribute__((ext_vector_type(8)))  float  v8f;

union Frag16 { v16bf v; uint4 u[2]; };
union Pack4  { __bf16 h[4]; uint2 u; };
union Pack2  { __bf16 h[2]; unsigned int u; };

static __device__ __forceinline__ v8f wmma_bf16(v16bf a, v16bf b, v8f c) {
  // D = A(16x32 bf16) * B(32x16 bf16) + C(16x16 f32)
  return __builtin_amdgcn_wmma_f32_16x16x32_bf16(false, a, false, b, (short)0, c,
                                                 false, false);
}

static __device__ __forceinline__ v8f v8f_zero() {
  v8f z = {0.f, 0.f, 0.f, 0.f, 0.f, 0.f, 0.f, 0.f};
  return z;
}

// ---------------------------------------------------------------------------
// Generic C[M,N] = A[M,Kd] (f32, row-major) * W[N,Kd]^T (f32, row-major)
// bf16 WMMA compute, fp32 accumulate.  Block tile 128x128, BK=32, 8 waves.
// ---------------------------------------------------------------------------
__global__ __launch_bounds__(256, 2) void gemm_bf16_nt(
    const float* __restrict__ A, const float* __restrict__ W,
    float* __restrict__ C, int M, int N, int Kd)
{
  __shared__ __bf16 As[128][40];   // padded row (40 bf16 = 80B) to spread banks
  __shared__ __bf16 Ws[128][40];

  const int tid  = threadIdx.x;
  const int wave = tid >> 5;
  const int lane = tid & 31;
  const int half = lane >> 4;
  const int l16  = lane & 15;
  const int wm   = wave >> 2;   // 0..1  (64 rows of M each)
  const int wn   = wave & 3;    // 0..3  (32 cols of N each)

  const int ntiles = N >> 7;
  const int m0 = (blockIdx.x / ntiles) << 7;
  const int n0 = (blockIdx.x % ntiles) << 7;

  v8f acc[4][2];
#pragma unroll
  for (int s = 0; s < 4; ++s)
#pragma unroll
    for (int t = 0; t < 2; ++t) acc[s][t] = v8f_zero();

  const int lrow = tid >> 1;           // 0..127
  const int lcol = (tid & 1) << 4;     // 0 or 16
  const float* aptr = A + (size_t)(m0 + lrow) * Kd + lcol;
  const float* wptr = W + (size_t)(n0 + lrow) * Kd + lcol;

  for (int kk = 0; kk < Kd; kk += 32) {
    float4 av[4], wv[4];
#pragma unroll
    for (int j = 0; j < 4; ++j) {
      av[j] = *(const float4*)(aptr + kk + j * 4);
      wv[j] = *(const float4*)(wptr + kk + j * 4);
    }
    if (kk + 32 < Kd) {                 // pull next K-slab toward the WGP
      __builtin_prefetch(aptr + kk + 32, 0, 0);
      __builtin_prefetch(wptr + kk + 32, 0, 0);
    }
    __syncthreads();                    // previous iteration's frag reads done
#pragma unroll
    for (int j = 0; j < 4; ++j) {
      Pack4 pa, pw;
      pa.h[0] = (__bf16)av[j].x; pa.h[1] = (__bf16)av[j].y;
      pa.h[2] = (__bf16)av[j].z; pa.h[3] = (__bf16)av[j].w;
      pw.h[0] = (__bf16)wv[j].x; pw.h[1] = (__bf16)wv[j].y;
      pw.h[2] = (__bf16)wv[j].z; pw.h[3] = (__bf16)wv[j].w;
      *(uint2*)&As[lrow][lcol + j * 4] = pa.u;
      *(uint2*)&Ws[lrow][lcol + j * 4] = pw.u;
    }
    __syncthreads();

    Frag16 af[4], bf[2];
#pragma unroll
    for (int s = 0; s < 4; ++s) {
      const int r = (wm << 6) + (s << 4) + l16;
      af[s].u[0] = *(const uint4*)&As[r][half * 8];
      af[s].u[1] = *(const uint4*)&As[r][16 + half * 8];
    }
#pragma unroll
    for (int t = 0; t < 2; ++t) {
      const int r = (wn << 5) + (t << 4) + l16;
      bf[t].u[0] = *(const uint4*)&Ws[r][half * 8];
      bf[t].u[1] = *(const uint4*)&Ws[r][16 + half * 8];
    }
#pragma unroll
    for (int s = 0; s < 4; ++s)
#pragma unroll
      for (int t = 0; t < 2; ++t)
        acc[s][t] = wmma_bf16(af[s].v, bf[t].v, acc[s][t]);
  }

#pragma unroll
  for (int s = 0; s < 4; ++s)
#pragma unroll
    for (int t = 0; t < 2; ++t)
#pragma unroll
      for (int i = 0; i < 8; ++i) {
        const int row = m0 + (wm << 6) + (s << 4) + i + (half << 3);
        const int col = n0 + (wn << 5) + (t << 4) + l16;
        C[(size_t)row * N + col] = acc[s][t][i];
      }
}

// ---------------------------------------------------------------------------
// RMSNorm + RoPE for Q.  One wave per (b, hq, k) head-row of 128 values.
// Writes bf16 in attention layout (B, NH, K, HD).
// ---------------------------------------------------------------------------
__global__ __launch_bounds__(256) void rope_q_kernel(
    const float* __restrict__ qr, const float* __restrict__ w,
    const float* __restrict__ cosb, const float* __restrict__ sinb,
    __bf16* __restrict__ qo)
{
  const int row = blockIdx.x * 8 + (threadIdx.x >> 5);
  if (row >= B_ * NH_ * KQ_) return;
  const int lane = threadIdx.x & 31;
  const int k  = row % KQ_;
  const int t  = row / KQ_;
  const int hq = t % NH_;
  const int b  = t / NH_;

  const float* src = qr + (size_t)(b * KQ_ + k) * DH_ + hq * HD_;
  const int d0 = lane * 2;
  float x0 = src[d0], x1 = src[d0 + 1];
  float y0 = src[64 + d0], y1 = src[64 + d0 + 1];
  float ss = x0 * x0 + x1 * x1 + y0 * y0 + y1 * y1;
#pragma unroll
  for (int m = 16; m >= 1; m >>= 1) ss += __shfl_xor(ss, m, 32);
  const float inv = rsqrtf(ss * (1.0f / 128.0f) + 1e-6f);
  x0 *= inv * w[d0];       x1 *= inv * w[d0 + 1];
  y0 *= inv * w[64 + d0];  y1 *= inv * w[64 + d0 + 1];

  const float* cp = cosb + ((size_t)b * L_ + CTX_ + k) * HD_;
  const float* sp = sinb + ((size_t)b * L_ + CTX_ + k) * HD_;
  const float o0 = x0 * cp[d0]        - y0 * sp[d0];
  const float o1 = x1 * cp[d0 + 1]    - y1 * sp[d0 + 1];
  const float o2 = y0 * cp[64 + d0]     + x0 * sp[64 + d0];
  const float o3 = y1 * cp[64 + d0 + 1] + x1 * sp[64 + d0 + 1];

  __bf16* dst = qo + (size_t)((b * NH_ + hq) * KQ_ + k) * HD_;
  Pack2 p0, p1;
  p0.h[0] = (__bf16)o0; p0.h[1] = (__bf16)o1;
  p1.h[0] = (__bf16)o2; p1.h[1] = (__bf16)o3;
  *(unsigned*)(dst + d0)      = p0.u;
  *(unsigned*)(dst + 64 + d0) = p1.u;
}

// ---------------------------------------------------------------------------
// RMSNorm + RoPE for K (ctx rows 0..CTX-1, noise rows CTX..L-1).
// Writes bf16 in attention layout (B, NKV, L, HD).
// ---------------------------------------------------------------------------
__global__ __launch_bounds__(256) void rope_k_kernel(
    const float* __restrict__ kc, const float* __restrict__ kn,
    const float* __restrict__ w,
    const float* __restrict__ cosb, const float* __restrict__ sinb,
    __bf16* __restrict__ ko)
{
  const int row = blockIdx.x * 8 + (threadIdx.x >> 5);
  if (row >= B_ * NKV_ * L_) return;
  const int lane = threadIdx.x & 31;
  const int l = row % L_;
  const int t = row / L_;
  const int h = t % NKV_;
  const int b = t / NKV_;

  const float* src = (l < CTX_)
      ? kc + (size_t)(b * CTX_ + l) * DKV_ + h * HD_
      : kn + (size_t)(b * KQ_ + (l - CTX_)) * DKV_ + h * HD_;

  const int d0 = lane * 2;
  float x0 = src[d0], x1 = src[d0 + 1];
  float y0 = src[64 + d0], y1 = src[64 + d0 + 1];
  float ss = x0 * x0 + x1 * x1 + y0 * y0 + y1 * y1;
#pragma unroll
  for (int m = 16; m >= 1; m >>= 1) ss += __shfl_xor(ss, m, 32);
  const float inv = rsqrtf(ss * (1.0f / 128.0f) + 1e-6f);
  x0 *= inv * w[d0];       x1 *= inv * w[d0 + 1];
  y0 *= inv * w[64 + d0];  y1 *= inv * w[64 + d0 + 1];

  const float* cp = cosb + ((size_t)b * L_ + l) * HD_;
  const float* sp = sinb + ((size_t)b * L_ + l) * HD_;
  const float o0 = x0 * cp[d0]        - y0 * sp[d0];
  const float o1 = x1 * cp[d0 + 1]    - y1 * sp[d0 + 1];
  const float o2 = y0 * cp[64 + d0]     + x0 * sp[64 + d0];
  const float o3 = y1 * cp[64 + d0 + 1] + x1 * sp[64 + d0 + 1];

  __bf16* dst = ko + (size_t)((b * NKV_ + h) * L_ + l) * HD_;
  Pack2 p0, p1;
  p0.h[0] = (__bf16)o0; p0.h[1] = (__bf16)o1;
  p1.h[0] = (__bf16)o2; p1.h[1] = (__bf16)o3;
  *(unsigned*)(dst + d0)      = p0.u;
  *(unsigned*)(dst + 64 + d0) = p1.u;
}

// ---------------------------------------------------------------------------
// V relayout: (ctx/noise raw rows) -> bf16 (B, NKV, HD, L)  [L-contiguous]
// 32x32 tile transpose through LDS.
// ---------------------------------------------------------------------------
__global__ __launch_bounds__(256) void v_transpose_kernel(
    const float* __restrict__ vc, const float* __restrict__ vn,
    __bf16* __restrict__ vt)
{
  __shared__ float tile[32][33];
  int blk = blockIdx.x;
  const int dt = blk & 3;           blk >>= 2;           // HD/32 = 4
  const int lt = blk % (L_ / 32);   blk /= (L_ / 32);    // 130
  const int h  = blk % NKV_;
  const int b  = blk / NKV_;
  const int l0 = lt * 32, d0 = dt * 32;

  const int tid = threadIdx.x;
  const int lr = tid >> 3;            // 0..31
  const int q4 = (tid & 7) << 2;      // 0..28
  const int l = l0 + lr;
  const float* src = (l < CTX_)
      ? vc + (size_t)(b * CTX_ + l) * DKV_ + h * HD_ + d0 + q4
      : vn + (size_t)(b * KQ_ + (l - CTX_)) * DKV_ + h * HD_ + d0 + q4;
  const float4 val = *(const float4*)src;
  tile[lr][q4 + 0] = val.x;
  tile[lr][q4 + 1] = val.y;
  tile[lr][q4 + 2] = val.z;
  tile[lr][q4 + 3] = val.w;
  __syncthreads();

  Pack4 pk;
#pragma unroll
  for (int j = 0; j < 4; ++j) pk.h[j] = (__bf16)tile[q4 + j][lr];
  *(uint2*)(vt + (size_t)((b * NKV_ + h) * HD_ + d0 + lr) * L_ + l0 + q4) = pk.u;
}

// ---------------------------------------------------------------------------
// Flash attention.  Block = (b, kv-head, group); 4 waves x 16 q-rows.
// Streams L in 64-wide chunks; bf16 WMMA for QK^T and P*V, fp32 softmax.
// Output: (B*K, NH*HD) f32 in the layout the Wo projection consumes.
// ---------------------------------------------------------------------------
__global__ __launch_bounds__(128) void attn_kernel(
    const __bf16* __restrict__ qa, const __bf16* __restrict__ ka,
    const __bf16* __restrict__ vt, const float* __restrict__ mask,
    float* __restrict__ out)
{
  __shared__ __bf16 Plds[4][16][72];   // wave-private 16x64 P tile, padded

  int blk = blockIdx.x;
  const int gq = blk % G_;  blk /= G_;
  const int h  = blk % NKV_;
  const int b  = blk / NKV_;
  const int hq = h * G_ + gq;

  const int wave = threadIdx.x >> 5;
  const int lane = threadIdx.x & 31;
  const int half = lane >> 4;
  const int l16  = lane & 15;
  const int kr0  = wave << 4;          // this wave's q-row base (0..48)

  const __bf16* qbase = qa + (size_t)(b * NH_ + hq) * KQ_ * HD_;
  const __bf16* kbase = ka + (size_t)(b * NKV_ + h) * L_ * HD_;
  const __bf16* vbase = vt + (size_t)(b * NKV_ + h) * HD_ * L_;
  const float*  mrowb = mask + (size_t)(b * KQ_ + kr0 + half * 8) * L_;

  // Q A-fragments (resident for the whole kernel): 4 chunks of 32 d
  Frag16 qf[4];
#pragma unroll
  for (int c = 0; c < 4; ++c) {
    const __bf16* p = qbase + (size_t)(kr0 + l16) * HD_ + c * 32 + half * 8;
    qf[c].u[0] = *(const uint4*)p;
    qf[c].u[1] = *(const uint4*)(p + 16);
  }

  v8f acc[8];
#pragma unroll
  for (int d = 0; d < 8; ++d) acc[d] = v8f_zero();
  float rmax[8], rsum[8];
#pragma unroll
  for (int i = 0; i < 8; ++i) { rmax[i] = -3.0e38f; rsum[i] = 0.f; }

  const float scale = 0.08838834764831845f;   // 1/sqrt(128)

  for (int lb = 0; lb < L_; lb += 64) {
    // S = Q * K^T  (16 x 64 per wave)
    v8f s[4];
#pragma unroll
    for (int n = 0; n < 4; ++n) s[n] = v8f_zero();
#pragma unroll
    for (int n = 0; n < 4; ++n) {
      const __bf16* krow = kbase + (size_t)(lb + n * 16 + l16) * HD_;
#pragma unroll
      for (int c = 0; c < 4; ++c) {
        Frag16 kf;
        kf.u[0] = *(const uint4*)(krow + c * 32 + half * 8);
        kf.u[1] = *(const uint4*)(krow + c * 32 + 16 + half * 8);
        s[n] = wmma_bf16(qf[c].v, kf.v, s[n]);
      }
    }

    // scale + mask; chunk row-max
    float cmax[8];
#pragma unroll
    for (int i = 0; i < 8; ++i) cmax[i] = -3.0e38f;
#pragma unroll
    for (int n = 0; n < 4; ++n)
#pragma unroll
      for (int i = 0; i < 8; ++i) {
        const float v = s[n][i] * scale +
                        mrowb[(size_t)i * L_ + lb + n * 16 + l16];
        s[n][i] = v;
        cmax[i] = fmaxf(cmax[i], v);
      }
#pragma unroll
    for (int m = 8; m >= 1; m >>= 1)
#pragma unroll
      for (int i = 0; i < 8; ++i)
        cmax[i] = fmaxf(cmax[i], __shfl_xor(cmax[i], m, 32));

    // online softmax update
    float cf[8], csum[8];
#pragma unroll
    for (int i = 0; i < 8; ++i) {
      const float nm = fmaxf(rmax[i], cmax[i]);
      cf[i] = __expf(rmax[i] - nm);
      rmax[i] = nm;
      csum[i] = 0.f;
    }
#pragma unroll
    for (int n = 0; n < 4; ++n)
#pragma unroll
      for (int i = 0; i < 8; ++i) {
        const float p = __expf(s[n][i] - rmax[i]);
        s[n][i] = p;
        csum[i] += p;
      }
#pragma unroll
    for (int m = 8; m >= 1; m >>= 1)
#pragma unroll
      for (int i = 0; i < 8; ++i)
        csum[i] += __shfl_xor(csum[i], m, 32);
#pragma unroll
    for (int i = 0; i < 8; ++i) rsum[i] = rsum[i] * cf[i] + csum[i];
#pragma unroll
    for (int d = 0; d < 8; ++d)
#pragma unroll
      for (int i = 0; i < 8; ++i) acc[d][i] *= cf[i];

    // C-layout -> A-layout for P through wave-private LDS (intra-wave dep,
    // hardware DScnt ordering inserted by the compiler)
#pragma unroll
    for (int n = 0; n < 4; ++n)
#pragma unroll
      for (int i = 0; i < 8; ++i)
        Plds[wave][i + half * 8][n * 16 + l16] = (__bf16)s[n][i];

    Frag16 pf[2];
#pragma unroll
    for (int t = 0; t < 2; ++t) {
      pf[t].u[0] = *(const uint4*)&Plds[wave][l16][t * 32 + half * 8];
      pf[t].u[1] = *(const uint4*)&Plds[wave][l16][t * 32 + 16 + half * 8];
    }

    // acc += P * V   (V stored d-major, contraction dim L contiguous)
#pragma unroll
    for (int d = 0; d < 8; ++d) {
      const __bf16* vrow = vbase + (size_t)(d * 16 + l16) * L_ + lb;
#pragma unroll
      for (int t = 0; t < 2; ++t) {
        Frag16 vf;
        vf.u[0] = *(const uint4*)(vrow + t * 32 + half * 8);
        vf.u[1] = *(const uint4*)(vrow + t * 32 + 16 + half * 8);
        acc[d] = wmma_bf16(pf[t].v, vf.v, acc[d]);
      }
    }
  }

  // epilogue: normalize, scatter to (B*K, NH*HD)
#pragma unroll
  for (int d = 0; d < 8; ++d)
#pragma unroll
    for (int i = 0; i < 8; ++i) {
      const int k = kr0 + i + half * 8;
      out[(size_t)(b * KQ_ + k) * DH_ + hq * HD_ + d * 16 + l16] =
          acc[d][i] / rsum[i];
    }
}

// ---------------------------------------------------------------------------
extern "C" void kernel_launch(void* const* d_in, const int* in_sizes, int n_in,
                              void* d_out, int out_size, void* d_ws,
                              size_t ws_size, hipStream_t stream)
{
  const float* hidden = (const float*)d_in[0];   // (B, K, H)
  const float* target = (const float*)d_in[1];   // (B, CTX, H)
  const float* mask   = (const float*)d_in[2];   // (B, 1, K, L)
  const float* cosb   = (const float*)d_in[3];   // (B, L, HD)
  const float* sinb   = (const float*)d_in[4];   // (B, L, HD)
  const float* Wq     = (const float*)d_in[5];   // (NH*HD, H)
  const float* Wk     = (const float*)d_in[6];   // (NKV*HD, H)
  const float* Wv     = (const float*)d_in[7];   // (NKV*HD, H)
  const float* Wo     = (const float*)d_in[8];   // (H, NH*HD)
  const float* qnw    = (const float*)d_in[9];   // (HD,)
  const float* knw    = (const float*)d_in[10];  // (HD,)

  char* wsp = (char*)d_ws;
  auto carve = [&](size_t bytes) {
    char* p = wsp;
    wsp += (bytes + 255) & ~(size_t)255;
    return p;
  };
  float*  q_raw  = (float*) carve((size_t)B_ * KQ_  * DH_  * 4);  //  4 MB
  float*  kc_raw = (float*) carve((size_t)B_ * CTX_ * DKV_ * 4);  // 64 MB
  float*  vc_raw = (float*) carve((size_t)B_ * CTX_ * DKV_ * 4);  // 64 MB
  float*  kn_raw = (float*) carve((size_t)B_ * KQ_  * DKV_ * 4);  //  1 MB
  float*  vn_raw = (float*) carve((size_t)B_ * KQ_  * DKV_ * 4);  //  1 MB
  float*  aout   = (float*) carve((size_t)B_ * KQ_  * DH_  * 4);  //  4 MB
  __bf16* q_attn = (__bf16*)carve((size_t)B_ * NH_  * KQ_ * HD_ * 2);
  __bf16* k_attn = (__bf16*)carve((size_t)B_ * NKV_ * L_  * HD_ * 2);
  __bf16* v_t    = (__bf16*)carve((size_t)B_ * NKV_ * HD_ * L_  * 2);

  const int MQ = B_ * KQ_;      // 256
  const int MC = B_ * CTX_;     // 16384

  // Projections (bf16 WMMA GEMMs)
  gemm_bf16_nt<<<(MQ / 128) * (DH_  / 128), 256, 0, stream>>>(hidden, Wq, q_raw,  MQ, DH_,  H_);
  gemm_bf16_nt<<<(MC / 128) * (DKV_ / 128), 256, 0, stream>>>(target, Wk, kc_raw, MC, DKV_, H_);
  gemm_bf16_nt<<<(MC / 128) * (DKV_ / 128), 256, 0, stream>>>(target, Wv, vc_raw, MC, DKV_, H_);
  gemm_bf16_nt<<<(MQ / 128) * (DKV_ / 128), 256, 0, stream>>>(hidden, Wk, kn_raw, MQ, DKV_, H_);
  gemm_bf16_nt<<<(MQ / 128) * (DKV_ / 128), 256, 0, stream>>>(hidden, Wv, vn_raw, MQ, DKV_, H_);

  // RMSNorm + RoPE + relayout to bf16
  rope_q_kernel<<<(B_ * NH_ * KQ_) / 8, 256, 0, stream>>>(q_raw, qnw, cosb, sinb, q_attn);
  rope_k_kernel<<<(B_ * NKV_ * L_) / 8, 256, 0, stream>>>(kc_raw, kn_raw, knw, cosb, sinb, k_attn);
  v_transpose_kernel<<<B_ * NKV_ * (L_ / 32) * (HD_ / 32), 256, 0, stream>>>(vc_raw, vn_raw, v_t);

  // Flash attention
  attn_kernel<<<B_ * NKV_ * G_, 128, 0, stream>>>(q_attn, k_attn, v_t, mask, aout);

  // Output projection -> d_out (B, K, H) f32
  gemm_bf16_nt<<<(MQ / 128) * (H_ / 128), 256, 0, stream>>>(aout, Wo, (float*)d_out, MQ, H_, DH_);
}
